// RelationModel_side_double_38259568673082
// MI455X (gfx1250) — compile-verified, matrix operand order checked by reading, not run on hardware
//
#include <hip/hip_runtime.h>
#include <hip/hip_bf16.h>

typedef __attribute__((ext_vector_type(16))) __bf16 bf16x16;
typedef __attribute__((ext_vector_type(8)))  float  f32x8;

#define BATCH 8
#define PSEQ 3136
#define DIMN 256
#define HEADS 8
#define DHEAD 32
#define SCALE_QK 0.17677669529663687f

// per-wave LDS layout (bytes)
#define Z_STRIDE 272                    // bf16 elems per row (16 elem pad)
#define Z_BYTES (16 * Z_STRIDE * 2)     // 8704
#define AO_BYTES (16 * Z_STRIDE * 2)    // 8704
#define QKV_STRIDE 100                  // f32 elems per row (400B, 16B aligned)
#define QKV_BYTES (16 * QKV_STRIDE * 4) // 6400
#define PER_WAVE (Z_BYTES + AO_BYTES + QKV_BYTES) // 23808 (mult of 32)
#define WAVES 8
#define SMEM_TOTAL (PER_WAVE * WAVES)   // 190464

__constant__ float c_masks[2][16] = {
    {   0.f,    0.f,    0.f, -100.f,
        0.f,    0.f, -100.f,    0.f,
        0.f, -100.f,    0.f,    0.f,
     -100.f,    0.f,    0.f,    0.f},
    {   0.f, -100.f,    0.f, -100.f,
     -100.f,    0.f, -100.f,    0.f,
     -100.f, -100.f, -100.f, -100.f,
     -100.f, -100.f, -100.f, -100.f}};

__device__ __forceinline__ float dot4(float4 a, float4 b) {
  return fmaf(a.x, b.x, fmaf(a.y, b.y, fmaf(a.z, b.z, a.w * b.w)));
}

// ---------------------------------------------------------------------------
// Kernel 0: saliency-thresholded global mean over P  (rgb_g, dep_g : [B][256])
// ---------------------------------------------------------------------------
__global__ __launch_bounds__(256) void saliency_reduce(
    const float* __restrict__ x, const float* __restrict__ y,
    const float* __restrict__ s, float* __restrict__ rgb_g,
    float* __restrict__ dep_g) {
  const int b = blockIdx.x;
  const int tid = threadIdx.x;
  __shared__ float red[256];

  float cnt = 0.f;
  for (int p = tid; p < PSEQ; p += 256)
    cnt += (s[b * PSEQ + p] > 0.5f) ? 1.0f : 1e-8f;
  red[tid] = cnt;
  __syncthreads();
  for (int off = 128; off > 0; off >>= 1) {
    if (tid < off) red[tid] += red[tid + off];
    __syncthreads();
  }
  const float inv_denom = 1.0f / (red[0] + 1e-8f);

  const int d = tid;  // 256 channels
  float ax = 0.f, ay = 0.f;
  for (int p = 0; p < PSEQ; ++p) {
    const float st = (s[b * PSEQ + p] > 0.5f) ? 1.0f : 1e-8f;
    const size_t base = ((size_t)(b * PSEQ + p)) * DIMN + d;
    ax = fmaf(x[base], st, ax);
    ay = fmaf(y[base], st, ay);
  }
  rgb_g[b * DIMN + d] = ax * inv_denom;
  dep_g[b * DIMN + d] = ay * inv_denom;
}

// ---------------------------------------------------------------------------
// Kernel 1: convert f32 weight [K=256][N] to bf16 in WMMA B-fragment order:
//   dst[((nt*8 + k)*32 + lane)*16 + i] = W[k*32 + (lane>>4)*16 + i][nt*16 + (lane&15)]
// so each lane's 16 bf16 values of a 32x16 B-tile are contiguous (32B/lane).
// ---------------------------------------------------------------------------
__global__ __launch_bounds__(256) void swizzle_weight(
    const float* __restrict__ W, __bf16* __restrict__ dst, int N, int total) {
  int idx = blockIdx.x * 256 + threadIdx.x;
  if (idx >= total) return;
  const int i = idx & 15;
  const int lane = (idx >> 4) & 31;
  const int k = (idx >> 9) & 7;
  const int nt = idx >> 12;
  const int krow = k * 32 + ((lane >> 4) << 4) + i;
  const int ncol = nt * 16 + (lane & 15);
  dst[idx] = (__bf16)W[(size_t)krow * N + ncol];
}

// ---------------------------------------------------------------------------
// Kernel 2: fully fused double attention block. One wave = one 16-row tile
// = 4 p-positions x 4 tokens. All LDS is per-wave private -> no barriers.
// ---------------------------------------------------------------------------
__global__ __launch_bounds__(256) void fused_double_attn(
    const float* __restrict__ x, const float* __restrict__ y,
    const float* __restrict__ rgb_g, const float* __restrict__ dep_g,
    const __bf16* __restrict__ wqkv0, const __bf16* __restrict__ wout0,
    const float* __restrict__ bout0, const __bf16* __restrict__ wqkv1,
    const __bf16* __restrict__ wout1, const float* __restrict__ bout1,
    float* __restrict__ out) {
  extern __shared__ __align__(128) char smem[];
  const int wid = threadIdx.x >> 5;
  const int lane = threadIdx.x & 31;

  char* base = smem + wid * PER_WAVE;
  __bf16* Z = (__bf16*)base;                 // [16][Z_STRIDE] bf16
  __bf16* AO = (__bf16*)(base + Z_BYTES);    // [16][Z_STRIDE] bf16
  float* QKV = (float*)(base + Z_BYTES + AO_BYTES);  // [16][QKV_STRIDE] f32

  const int tile = blockIdx.x * WAVES + wid;     // 6272 tiles total
  const int b = tile / (PSEQ / 4);
  const int p0 = (tile % (PSEQ / 4)) * 4;

  // ---- build input tile Z: rows r = 4*gi + t ------------------------------
  // each lane owns a contiguous 32B chunk of the 1KB f32 row (b128 coalesced)
  for (int r = 0; r < 16; ++r) {
    const int t = r & 3, gi = r >> 2;
    const float* src;
    if (t == 0)      src = x + ((size_t)(b * PSEQ + p0 + gi)) * DIMN;
    else if (t == 1) src = y + ((size_t)(b * PSEQ + p0 + gi)) * DIMN;
    else if (t == 2) src = rgb_g + b * DIMN;
    else             src = dep_g + b * DIMN;
    const float4 v0 = ((const float4*)src)[lane * 2 + 0];
    const float4 v1 = ((const float4*)src)[lane * 2 + 1];
    __bf16* zr = Z + r * Z_STRIDE + lane * 8;
    zr[0] = (__bf16)v0.x; zr[1] = (__bf16)v0.y;
    zr[2] = (__bf16)v0.z; zr[3] = (__bf16)v0.w;
    zr[4] = (__bf16)v1.x; zr[5] = (__bf16)v1.y;
    zr[6] = (__bf16)v1.z; zr[7] = (__bf16)v1.w;
  }

  // lane roles for the scalar 4x4 attention core
  const int dhalf = lane & 1;          // which half of DHEAD
  const int ti = (lane >> 1) & 3;      // token i within group
  const int gg = lane >> 3;            // group 0..3

  // C-layout scatter helpers
  const int row0 = (lane >> 4) * 8;    // lanes 16-31 hold M = v+8
  const int nn = lane & 15;            // column within N-tile

  for (int blk = 0; blk < 2; ++blk) {
    const __bf16* wqkv = blk ? wqkv1 : wqkv0;
    const __bf16* wout = blk ? wout1 : wout0;
    const float* bout = blk ? bout1 : bout0;
    const float* mask = c_masks[blk];

    for (int h = 0; h < HEADS; ++h) {
      // ---- QKV projection for head h: 6 N-tiles x 8 K-steps of WMMA ------
#pragma unroll
      for (int part = 0; part < 3; ++part) {    // 0=q 1=k 2=v
#pragma unroll
        for (int nt2 = 0; nt2 < 2; ++nt2) {
          const int nt = part * 16 + h * 2 + nt2;
          f32x8 acc = {};
#pragma unroll
          for (int k = 0; k < 8; ++k) {
            bf16x16 a = *(const bf16x16*)(Z + (lane & 15) * Z_STRIDE +
                                          k * 32 + (lane >> 4) * 16);
            bf16x16 bb = *(const bf16x16*)(wqkv +
                         ((size_t)((nt * 8 + k) * 32 + lane)) * 16);
            acc = __builtin_amdgcn_wmma_f32_16x16x32_bf16(
                false, a, false, bb, (short)0, acc, false, false);
          }
          const int c = part * 32 + nt2 * 16 + nn;
#pragma unroll
          for (int v = 0; v < 8; ++v)
            QKV[(row0 + v) * QKV_STRIDE + c] = acc[v];
        }
      }

      // ---- scalar 4x4 softmax attention (lane = (g, i, dhalf)) -----------
      // rows are 16B aligned -> all LDS reads are ds_load_b128
      {
        const int rb = gg * 4;
        const float4* qrow = (const float4*)(QKV + (rb + ti) * QKV_STRIDE);
        float4 q[8];
#pragma unroll
        for (int c = 0; c < 8; ++c) q[c] = qrow[c];

        float dots[4];
#pragma unroll
        for (int j = 0; j < 4; ++j) {
          const float4* krow =
              (const float4*)(QKV + (rb + j) * QKV_STRIDE + 32);
          float d = 0.f;
#pragma unroll
          for (int c = 0; c < 8; ++c) d += dot4(q[c], krow[c]);
          dots[j] = d * SCALE_QK + mask[ti * 4 + j];
        }
        const float m =
            fmaxf(fmaxf(dots[0], dots[1]), fmaxf(dots[2], dots[3]));
        const float e0 = __expf(dots[0] - m), e1 = __expf(dots[1] - m);
        const float e2 = __expf(dots[2] - m), e3 = __expf(dots[3] - m);
        const float inv = 1.0f / (e0 + e1 + e2 + e3);

        const int vcol = 64 + dhalf * 16;  // 16 f32 of V per lane
        const float4* v0r = (const float4*)(QKV + (rb + 0) * QKV_STRIDE + vcol);
        const float4* v1r = (const float4*)(QKV + (rb + 1) * QKV_STRIDE + vcol);
        const float4* v2r = (const float4*)(QKV + (rb + 2) * QKV_STRIDE + vcol);
        const float4* v3r = (const float4*)(QKV + (rb + 3) * QKV_STRIDE + vcol);
        __bf16* ao = AO + (rb + ti) * Z_STRIDE + h * DHEAD + dhalf * 16;
#pragma unroll
        for (int c = 0; c < 4; ++c) {
          const float4 a0 = v0r[c], a1 = v1r[c], a2 = v2r[c], a3 = v3r[c];
          ao[c * 4 + 0] = (__bf16)((e0 * a0.x + e1 * a1.x + e2 * a2.x + e3 * a3.x) * inv);
          ao[c * 4 + 1] = (__bf16)((e0 * a0.y + e1 * a1.y + e2 * a2.y + e3 * a3.y) * inv);
          ao[c * 4 + 2] = (__bf16)((e0 * a0.z + e1 * a1.z + e2 * a2.z + e3 * a3.z) * inv);
          ao[c * 4 + 3] = (__bf16)((e0 * a0.w + e1 * a1.w + e2 * a2.w + e3 * a3.w) * inv);
        }
      }
    }

    // ---- output projection: 16 N-tiles x 8 K-steps, bias in C ------------
    // unroll 2 -> two independent wmma chains to hide B-fragment latency
#pragma unroll 2
    for (int nt = 0; nt < 16; ++nt) {
      const float bias = bout[nt * 16 + nn];
      f32x8 acc;
#pragma unroll
      for (int v = 0; v < 8; ++v) acc[v] = bias;
#pragma unroll
      for (int k = 0; k < 8; ++k) {
        bf16x16 a = *(const bf16x16*)(AO + (lane & 15) * Z_STRIDE + k * 32 +
                                      (lane >> 4) * 16);
        bf16x16 bb = *(const bf16x16*)(wout +
                     ((size_t)((nt * 8 + k) * 32 + lane)) * 16);
        acc = __builtin_amdgcn_wmma_f32_16x16x32_bf16(
            false, a, false, bb, (short)0, acc, false, false);
      }
      if (blk == 0) {
        // feed block 2: overwrite Z tile (same-wave LDS is in-order)
#pragma unroll
        for (int v = 0; v < 8; ++v)
          Z[(row0 + v) * Z_STRIDE + nt * 16 + nn] = (__bf16)acc[v];
      } else {
        // final: only tokens 0 (x-stream) and 1 (y-stream) are outputs
#pragma unroll
        for (int v = 0; v < 8; ++v) {
          const int r = row0 + v;
          const int t = r & 3;
          if (t < 2) {
            const int gi = r >> 2;
            out[(size_t)t * ((size_t)BATCH * PSEQ * DIMN) +
                ((size_t)(b * PSEQ + p0 + gi)) * DIMN + nt * 16 + nn] = acc[v];
          }
        }
      }
    }
  }
}

// ---------------------------------------------------------------------------
extern "C" void kernel_launch(void* const* d_in, const int* in_sizes, int n_in,
                              void* d_out, int out_size, void* d_ws,
                              size_t ws_size, hipStream_t stream) {
  (void)in_sizes; (void)n_in; (void)out_size; (void)ws_size;
  const float* x = (const float*)d_in[0];
  const float* y = (const float*)d_in[1];
  const float* s = (const float*)d_in[2];
  const float* Wqkv0 = (const float*)d_in[3];
  const float* Wout0 = (const float*)d_in[4];
  const float* bout0 = (const float*)d_in[5];
  const float* Wqkv1 = (const float*)d_in[6];
  const float* Wout1 = (const float*)d_in[7];
  const float* bout1 = (const float*)d_in[8];
  float* out = (float*)d_out;

  // workspace partition (bytes)
  char* ws = (char*)d_ws;
  float* rgb_g = (float*)(ws + 0);                       //  8*256 f32
  float* dep_g = (float*)(ws + 8192);                    //  8*256 f32
  __bf16* wqkv0_s = (__bf16*)(ws + 16384);               // 196608 bf16
  __bf16* wout0_s = (__bf16*)(ws + 16384 + 393216);      //  65536 bf16
  __bf16* wqkv1_s = (__bf16*)(ws + 16384 + 393216 + 131072);
  __bf16* wout1_s = (__bf16*)(ws + 16384 + 2 * 393216 + 131072);

  const int NQKV = 256 * 768;  // 196608
  const int NOUT = 256 * 256;  //  65536

  swizzle_weight<<<(NQKV + 255) / 256, 256, 0, stream>>>(Wqkv0, wqkv0_s, 768, NQKV);
  swizzle_weight<<<(NOUT + 255) / 256, 256, 0, stream>>>(Wout0, wout0_s, 256, NOUT);
  swizzle_weight<<<(NQKV + 255) / 256, 256, 0, stream>>>(Wqkv1, wqkv1_s, 768, NQKV);
  swizzle_weight<<<(NOUT + 255) / 256, 256, 0, stream>>>(Wout1, wout1_s, 256, NOUT);
  saliency_reduce<<<BATCH, 256, 0, stream>>>(x, y, s, rgb_g, dep_g);

  const int tiles = BATCH * (PSEQ / 4);          // 6272
  const int blocks = tiles / WAVES;              // 784
  fused_double_attn<<<blocks, 256, SMEM_TOTAL, stream>>>(
      x, y, rgb_g, dep_g, wqkv0_s, wout0_s, bout0, wqkv1_s, wout1_s, bout1,
      out);
}
